// CausalSelfAttention_44547400794137
// MI455X (gfx1250) — matrix operand order repaired and checked by reading, not compiled, not run on hardware
//
#include <hip/hip_runtime.h>
#include <hip/hip_bf16.h>

// ---------------------------------------------------------------------------
// Problem constants (match reference)
// ---------------------------------------------------------------------------
#define BB   4
#define TT   2048
#define HID  1024
#define NH   16
#define HD   64
#define MROWS (BB * TT)          // 8192
#define NEGM (-10000.0f)

typedef __attribute__((ext_vector_type(16))) __bf16 v16bf;
typedef __attribute__((ext_vector_type(8)))  float  v8f;

// TDM availability / arity detection (bridge doc: clang-22 = 5 args, clang-23 = 6)
#if defined(__has_builtin)
#  if __has_builtin(__builtin_amdgcn_tensor_load_to_lds) && \
      __has_builtin(__builtin_amdgcn_s_wait_tensorcnt)
#    define USE_TDM 1
#  endif
#endif
#ifndef USE_TDM
#  define USE_TDM 0
#endif

union FragU {
    v16bf v;
    uint4 q[2];
};

static __device__ inline v16bf load_frag2(const unsigned short* p0,
                                          const unsigned short* p1) {
    FragU f;
    f.q[0] = *(const uint4*)p0;   // elements 0..7
    f.q[1] = *(const uint4*)p1;   // elements 8..15
    return f.v;
}

static __device__ inline unsigned short f32_to_bf16(float f) {
    unsigned int u = __float_as_uint(f);
    u += 0x7fffu + ((u >> 16) & 1u);   // round-to-nearest-even
    return (unsigned short)(u >> 16);
}

// ---------------------------------------------------------------------------
// fp32 -> bf16 conversion (vectorized, 4 elems/thread)
// ---------------------------------------------------------------------------
__global__ void cvt_bf16_kernel(const float* __restrict__ src,
                                unsigned short* __restrict__ dst, int n4) {
    int i = blockIdx.x * blockDim.x + threadIdx.x;
    if (i < n4) {
        float4 f = ((const float4*)src)[i];
        ushort4 o;
        o.x = f32_to_bf16(f.x);
        o.y = f32_to_bf16(f.y);
        o.z = f32_to_bf16(f.z);
        o.w = f32_to_bf16(f.w);
        ((ushort4*)dst)[i] = o;
    }
}

// ---------------------------------------------------------------------------
// Fused QKV projection:  y[m,n] = sum_k X[m,k] * W[n,k] + b[n]
// grid = (MROWS/64, HID/64, 3) ; block = 128 threads (4 waves)
// z==0 -> Q [B,NH,T,HD], z==1 -> K [B,NH,T,HD], z==2 -> V^T [B,NH,HD,T]
// ---------------------------------------------------------------------------
__global__ __launch_bounds__(128)
void qkv_proj_kernel(const unsigned short* __restrict__ Xb,
                     const unsigned short* __restrict__ Wqb,
                     const unsigned short* __restrict__ Wkb,
                     const unsigned short* __restrict__ Wvb,
                     const float* __restrict__ bq,
                     const float* __restrict__ bk,
                     const float* __restrict__ bv,
                     unsigned short* __restrict__ Qb,
                     unsigned short* __restrict__ Kb,
                     unsigned short* __restrict__ Vtb) {
    const int z = blockIdx.z;
    const unsigned short* W = (z == 0) ? Wqb : (z == 1) ? Wkb : Wvb;
    const float* bias       = (z == 0) ? bq  : (z == 1) ? bk  : bv;

    const int lane = threadIdx.x & 31;
    const int wave = threadIdx.x >> 5;
    const int l15  = lane & 15;
    const int h    = lane >> 4;

    const int m0 = blockIdx.x * 64 + (wave >> 1) * 32;
    const int n0 = blockIdx.y * 64 + (wave & 1) * 32;

    v8f acc[2][2];
#pragma unroll
    for (int i = 0; i < 2; ++i)
#pragma unroll
        for (int j = 0; j < 2; ++j) acc[i][j] = (v8f)0.0f;

    for (int k0 = 0; k0 < HID; k0 += 32) {
        v16bf afr[2], bfr[2];
#pragma unroll
        for (int s = 0; s < 2; ++s) {
            const unsigned short* arow = Xb + (size_t)(m0 + s * 16 + l15) * HID + k0;
            afr[s] = load_frag2(arow + h * 16, arow + h * 16 + 8);
            const unsigned short* brow = W + (size_t)(n0 + s * 16 + l15) * HID + k0 + h * 16;
            bfr[s] = load_frag2(brow, brow + 8);
        }
#pragma unroll
        for (int i = 0; i < 2; ++i)
#pragma unroll
            for (int j = 0; j < 2; ++j)
                acc[i][j] = __builtin_amdgcn_wmma_f32_16x16x32_bf16(
                    false, afr[i], false, bfr[j], (short)0, acc[i][j], false, false);
    }

    // bias epilogue (fp32)
#pragma unroll
    for (int j = 0; j < 2; ++j) {
        float bval = bias[n0 + j * 16 + l15];
#pragma unroll
        for (int i = 0; i < 2; ++i)
#pragma unroll
            for (int r = 0; r < 8; ++r) acc[i][j][r] += bval;
    }

    const int bsel = m0 >> 11;       // batch index (m0 / 2048)
    const int trow = m0 & (TT - 1);  // row within batch
    const int hsel = blockIdx.y;     // 64-wide N tile == one head

    if (z < 2) {
        unsigned short* dst = ((z == 0) ? Qb : Kb) + (size_t)(bsel * NH + hsel) * TT * HD;
#pragma unroll
        for (int i = 0; i < 2; ++i)
#pragma unroll
            for (int j = 0; j < 2; ++j) {
                int d = (wave & 1) * 32 + j * 16 + l15;
#pragma unroll
                for (int r = 0; r < 8; ++r) {
                    int t = trow + i * 16 + r + 8 * h;
                    dst[(size_t)t * HD + d] = f32_to_bf16(acc[i][j][r]);
                }
            }
    } else {
        // V transposed: [b,h][d][t] -> 8 consecutive t per lane = one 16B store
        unsigned short* dst = Vtb + (size_t)(bsel * NH + hsel) * HD * TT;
#pragma unroll
        for (int i = 0; i < 2; ++i)
#pragma unroll
            for (int j = 0; j < 2; ++j) {
                int d = (wave & 1) * 32 + j * 16 + l15;
                int tbase = trow + i * 16 + 8 * h;
                union { unsigned short u[8]; uint4 q; } pk;
#pragma unroll
                for (int r = 0; r < 8; ++r) pk.u[r] = f32_to_bf16(acc[i][j][r]);
                *(uint4*)(dst + (size_t)d * TT + tbase) = pk.q;
            }
    }
}

#if USE_TDM
// ---------------------------------------------------------------------------
// TDM helper: 2D tile load Global -> LDS.
// D# packed per CDNA5 ISA:
//   group0: count=1, lds_addr, global_addr(57b), type=2
//   group1: data_size=2B, tensor_dim = tile_dim (tile fully in bounds),
//           tensor_dim0_stride = row stride (elements)
// ---------------------------------------------------------------------------
typedef __attribute__((ext_vector_type(4))) unsigned int u32x4;
typedef __attribute__((ext_vector_type(8))) int          i32x8;
typedef __attribute__((ext_vector_type(4))) int          i32x4;

static __device__ inline void tdm_load_2d_bf16(unsigned int lds_byte_off,
                                               const unsigned short* gptr,
                                               unsigned int row_elems,
                                               unsigned int rows,
                                               unsigned long long row_stride) {
    unsigned long long ga = (unsigned long long)(size_t)gptr;
    union { unsigned int a[4]; u32x4 v; } g0;
    g0.a[0] = 1u;                                        // count=1 (user D#)
    g0.a[1] = lds_byte_off;                              // lds_addr (bytes)
    g0.a[2] = (unsigned int)(ga & 0xffffffffu);          // global_addr[31:0]
    g0.a[3] = (unsigned int)((ga >> 32) & 0x1ffffffu)    // global_addr[56:32]
            | 0x80000000u;                               // type = 2 ("image")
    union { unsigned int a[8]; i32x8 v; } g1;
    g1.a[0] = 1u << 16;                                  // data_size = 1 -> 2 bytes
    g1.a[1] = (row_elems & 0xffffu) << 16;               // tensor_dim0 lo
    g1.a[2] = (row_elems >> 16) | ((rows & 0xffffu) << 16);   // dim0 hi | tensor_dim1 lo
    g1.a[3] = (rows >> 16) | ((row_elems & 0xffffu) << 16);   // dim1 hi | tile_dim0
    g1.a[4] = rows & 0xffffu;                            // tile_dim1 (tile_dim2 = 0)
    g1.a[5] = (unsigned int)(row_stride & 0xffffffffu);  // tensor_dim0_stride lo
    g1.a[6] = (unsigned int)((row_stride >> 32) & 0xffffu);   // stride hi (dim1_stride = 0)
    g1.a[7] = 0u;
    i32x4 z4 = (i32x4)0;
#if defined(__clang_major__) && (__clang_major__ >= 23)
    i32x8 z8 = (i32x8)0;
    __builtin_amdgcn_tensor_load_to_lds(g0.v, g1.v, z4, z4, z8, 0);
#else
    __builtin_amdgcn_tensor_load_to_lds(g0.v, g1.v, z4, z4, 0);
#endif
}

// ---------------------------------------------------------------------------
// Flash attention, TDM double-buffered K/V staging shared by all 4 waves.
// grid = (T/64, B*NH) ; block = 128 threads (4 waves); wave owns 16 rows.
// All waves sweep the same block-level s-range (masked tiles underflow to
// exact fp32 zero, identical to the reference), enabling block barriers.
// ---------------------------------------------------------------------------
__global__ __launch_bounds__(128)
void attn_kernel(const unsigned short* __restrict__ Qb,
                 const unsigned short* __restrict__ Kb,
                 const unsigned short* __restrict__ Vtb,
                 const float* __restrict__ amask,
                 float* __restrict__ out) {
    __shared__ unsigned short kbuf[2][32 * 64];   // [s_local][d]
    __shared__ unsigned short vbuf[2][64 * 32];   // [d][s_local]
    __shared__ unsigned short pbuf[4][512];       // per-wave 16x32 P tile

    const int lane = threadIdx.x & 31;
    const int wave = threadIdx.x >> 5;
    const int l15  = lane & 15;
    const int h    = lane >> 4;

    const int bh    = blockIdx.y;       // b*NH + head
    const int bsel  = bh >> 4;
    const int head  = bh & 15;
    const int m0b   = blockIdx.x * 64;  // block row base within (b,h)
    const int m0    = m0b + wave * 16;  // wave row base

    const unsigned short* q  = Qb  + (size_t)bh * TT * HD;
    const unsigned short* kp = Kb  + (size_t)bh * TT * HD;
    const unsigned short* vt = Vtb + (size_t)bh * HD * TT;
    unsigned short* plds = &pbuf[wave][0];

    // Q fragments (d = 0..31, 32..63), resident for whole loop
    v16bf qf[2];
    {
        const unsigned short* qrow = q + (size_t)(m0 + l15) * HD;
        qf[0] = load_frag2(qrow + h * 16,      qrow + h * 16 + 8);
        qf[1] = load_frag2(qrow + 32 + h * 16, qrow + 40 + h * 16);
    }

    v8f o[4];
#pragma unroll
    for (int t = 0; t < 4; ++t) o[t] = (v8f)0.0f;
    float rm[8], rs[8];
#pragma unroll
    for (int r = 0; r < 8; ++r) { rm[r] = -1e30f; rs[r] = 0.0f; }

    // block-uniform column range; tiles fully inside the -1e4 window band
    // (every block row >= 513 has a zero-mask column) underflow to exact 0
    int s_hi = m0b + 64;
    if (m0b >= 576) {
        int cut = ((m0b + 63 - 512) + 31) & ~31;
        if (cut < s_hi) s_hi = cut;
    }
    const int steps = s_hi >> 5;

    // prologue: stage tile 0
    if (wave == 0) {
        tdm_load_2d_bf16((unsigned int)(size_t)&kbuf[0][0], kp, HD, 32, HD);
        tdm_load_2d_bf16((unsigned int)(size_t)&vbuf[0][0], vt, 32, HD, TT);
        __builtin_amdgcn_s_wait_tensorcnt((short)0);
    }
    __syncthreads();

    for (int it = 0; it < steps; ++it) {
        const int cur = it & 1;
        const int s0  = it << 5;

        // issue next tile's DMA while computing the current one
        if (wave == 0 && it + 1 < steps) {
            int sn = (it + 1) << 5;
            tdm_load_2d_bf16((unsigned int)(size_t)&kbuf[cur ^ 1][0],
                             kp + (size_t)sn * HD, HD, 32, HD);
            tdm_load_2d_bf16((unsigned int)(size_t)&vbuf[cur ^ 1][0],
                             vt + sn, 32, HD, TT);
        }

        // ---- S = Q @ K^T (two 16x16 column sub-tiles) from LDS ----
        v8f sc[2];
#pragma unroll
        for (int t = 0; t < 2; ++t) {
            sc[t] = (v8f)0.0f;
            const unsigned short* krow = &kbuf[cur][(t * 16 + l15) * HD + h * 16];
            v16bf kf0 = load_frag2(krow,      krow + 8);    // d 0..31
            v16bf kf1 = load_frag2(krow + 32, krow + 40);   // d 32..63
            sc[t] = __builtin_amdgcn_wmma_f32_16x16x32_bf16(
                false, qf[0], false, kf0, (short)0, sc[t], false, false);
            sc[t] = __builtin_amdgcn_wmma_f32_16x16x32_bf16(
                false, qf[1], false, kf1, (short)0, sc[t], false, false);
        }

        // ---- scale + mask (fp32, C layout: row = r+8h, col = l15) ----
        float am0 = amask[bsel * TT + s0 + l15];
        float am1 = amask[bsel * TT + s0 + 16 + l15];
#pragma unroll
        for (int t = 0; t < 2; ++t) {
            int sg = s0 + t * 16 + l15;
            float amv = t ? am1 : am0;
#pragma unroll
            for (int r = 0; r < 8; ++r) {
                int mg = m0 + r + 8 * h;
                float msk = ((sg > mg) ? NEGM : 0.0f) +
                            ((sg >= mg - 512) ? NEGM : 0.0f) + amv;
                sc[t][r] = sc[t][r] * 0.125f + msk;   // 1/sqrt(64)
            }
        }

        // ---- online softmax: row max over the 16 lanes of the half-wave ----
        float corr[8];
#pragma unroll
        for (int r = 0; r < 8; ++r) {
            float v = fmaxf(sc[0][r], sc[1][r]);
            v = fmaxf(v, __shfl_xor(v, 1, 16));
            v = fmaxf(v, __shfl_xor(v, 2, 16));
            v = fmaxf(v, __shfl_xor(v, 4, 16));
            v = fmaxf(v, __shfl_xor(v, 8, 16));
            float mnew = fmaxf(rm[r], v);
            corr[r] = __expf(rm[r] - mnew);
            rm[r] = mnew;
        }

        // ---- P = exp(S - max); stage bf16 P to LDS (C layout) ----
#pragma unroll
        for (int t = 0; t < 2; ++t)
#pragma unroll
            for (int r = 0; r < 8; ++r) {
                float p = __expf(sc[t][r] - rm[r]);
                sc[t][r] = p;
                plds[(r + 8 * h) * 32 + t * 16 + l15] = f32_to_bf16(p);
            }

        // ---- row sums ----
#pragma unroll
        for (int r = 0; r < 8; ++r) {
            float v = sc[0][r] + sc[1][r];
            v += __shfl_xor(v, 1, 16);
            v += __shfl_xor(v, 2, 16);
            v += __shfl_xor(v, 4, 16);
            v += __shfl_xor(v, 8, 16);
            rs[r] = rs[r] * corr[r] + v;
        }

        asm volatile("" ::: "memory");   // order P stores before A-layout reads

        // ---- re-read P as an A fragment (C->A relayout via LDS) ----
        const unsigned short* prow = plds + l15 * 32;
        v16bf pf = load_frag2(prow + h * 16, prow + h * 16 + 8);

        // ---- rescale running output, accumulate P @ V from LDS ----
#pragma unroll
        for (int t = 0; t < 4; ++t)
#pragma unroll
            for (int r = 0; r < 8; ++r) o[t][r] *= corr[r];

#pragma unroll
        for (int t = 0; t < 4; ++t) {
            const unsigned short* vrow = &vbuf[cur][(t * 16 + l15) * 32 + h * 16];
            v16bf vf = load_frag2(vrow, vrow + 8);
            o[t] = __builtin_amdgcn_wmma_f32_16x16x32_bf16(
                false, pf, false, vf, (short)0, o[t], false, false);
        }

        if (wave == 0) __builtin_amdgcn_s_wait_tensorcnt((short)0);
        __syncthreads();   // next-tile DMA visible; current buffers reusable
    }

    // ---- normalize and store fp32 output [B,T,NH,HD] ----
    float* op = out + (size_t)bsel * TT * HID + (size_t)head * HD;
#pragma unroll
    for (int t = 0; t < 4; ++t)
#pragma unroll
        for (int r = 0; r < 8; ++r) {
            int tt = m0 + r + 8 * h;
            op[(size_t)tt * HID + t * 16 + l15] = o[t][r] / rs[r];
        }
}

#else  // !USE_TDM: direct-global fallback (round-1 path)

__global__ __launch_bounds__(128)
void attn_kernel(const unsigned short* __restrict__ Qb,
                 const unsigned short* __restrict__ Kb,
                 const unsigned short* __restrict__ Vtb,
                 const float* __restrict__ amask,
                 float* __restrict__ out) {
    __shared__ unsigned short pbuf[4][512];

    const int lane = threadIdx.x & 31;
    const int wave = threadIdx.x >> 5;
    const int l15  = lane & 15;
    const int h    = lane >> 4;

    const int bh   = blockIdx.y;
    const int bsel = bh >> 4;
    const int head = bh & 15;
    const int m0   = blockIdx.x * 64 + wave * 16;

    const unsigned short* q  = Qb  + (size_t)bh * TT * HD;
    const unsigned short* kp = Kb  + (size_t)bh * TT * HD;
    const unsigned short* vt = Vtb + (size_t)bh * HD * TT;
    unsigned short* plds = &pbuf[wave][0];

    v16bf qf[2];
    {
        const unsigned short* qrow = q + (size_t)(m0 + l15) * HD;
        qf[0] = load_frag2(qrow + h * 16,      qrow + h * 16 + 8);
        qf[1] = load_frag2(qrow + 32 + h * 16, qrow + 40 + h * 16);
    }

    v8f o[4];
#pragma unroll
    for (int t = 0; t < 4; ++t) o[t] = (v8f)0.0f;
    float rm[8], rs[8];
#pragma unroll
    for (int r = 0; r < 8; ++r) { rm[r] = -1e30f; rs[r] = 0.0f; }

    const int s_end = m0 + 16;
    for (int s0 = 0; s0 < s_end; s0 += 32) {
        if (m0 >= 528 && s0 >= m0 + 15 - 512) break;

        v8f sc[2];
#pragma unroll
        for (int t = 0; t < 2; ++t) {
            sc[t] = (v8f)0.0f;
            const unsigned short* krow = kp + (size_t)(s0 + t * 16 + l15) * HD + h * 16;
            v16bf kf0 = load_frag2(krow,      krow + 8);
            v16bf kf1 = load_frag2(krow + 32, krow + 40);
            sc[t] = __builtin_amdgcn_wmma_f32_16x16x32_bf16(
                false, qf[0], false, kf0, (short)0, sc[t], false, false);
            sc[t] = __builtin_amdgcn_wmma_f32_16x16x32_bf16(
                false, qf[1], false, kf1, (short)0, sc[t], false, false);
        }

        float am0 = amask[bsel * TT + s0 + l15];
        float am1 = amask[bsel * TT + s0 + 16 + l15];
#pragma unroll
        for (int t = 0; t < 2; ++t) {
            int sg = s0 + t * 16 + l15;
            float amv = t ? am1 : am0;
#pragma unroll
            for (int r = 0; r < 8; ++r) {
                int mg = m0 + r + 8 * h;
                float msk = ((sg > mg) ? NEGM : 0.0f) +
                            ((sg >= mg - 512) ? NEGM : 0.0f) + amv;
                sc[t][r] = sc[t][r] * 0.125f + msk;
            }
        }

        float corr[8];
#pragma unroll
        for (int r = 0; r < 8; ++r) {
            float v = fmaxf(sc[0][r], sc[1][r]);
            v = fmaxf(v, __shfl_xor(v, 1, 16));
            v = fmaxf(v, __shfl_xor(v, 2, 16));
            v = fmaxf(v, __shfl_xor(v, 4, 16));
            v = fmaxf(v, __shfl_xor(v, 8, 16));
            float mnew = fmaxf(rm[r], v);
            corr[r] = __expf(rm[r] - mnew);
            rm[r] = mnew;
        }

#pragma unroll
        for (int t = 0; t < 2; ++t)
#pragma unroll
            for (int r = 0; r < 8; ++r) {
                float p = __expf(sc[t][r] - rm[r]);
                sc[t][r] = p;
                plds[(r + 8 * h) * 32 + t * 16 + l15] = f32_to_bf16(p);
            }

#pragma unroll
        for (int r = 0; r < 8; ++r) {
            float v = sc[0][r] + sc[1][r];
            v += __shfl_xor(v, 1, 16);
            v += __shfl_xor(v, 2, 16);
            v += __shfl_xor(v, 4, 16);
            v += __shfl_xor(v, 8, 16);
            rs[r] = rs[r] * corr[r] + v;
        }

        asm volatile("" ::: "memory");

        const unsigned short* prow = plds + l15 * 32;
        v16bf pf = load_frag2(prow + h * 16, prow + h * 16 + 8);

#pragma unroll
        for (int t = 0; t < 4; ++t)
#pragma unroll
            for (int r = 0; r < 8; ++r) o[t][r] *= corr[r];

#pragma unroll
        for (int t = 0; t < 4; ++t) {
            const unsigned short* vrow = vt + (size_t)(t * 16 + l15) * TT + s0 + h * 16;
            v16bf vf = load_frag2(vrow, vrow + 8);
            o[t] = __builtin_amdgcn_wmma_f32_16x16x32_bf16(
                false, pf, false, vf, (short)0, o[t], false, false);
        }
    }

    float* op = out + (size_t)bsel * TT * HID + (size_t)head * HD;
#pragma unroll
    for (int t = 0; t < 4; ++t)
#pragma unroll
        for (int r = 0; r < 8; ++r) {
            int tt = m0 + r + 8 * h;
            op[(size_t)tt * HID + t * 16 + l15] = o[t][r] / rs[r];
        }
}

#endif  // USE_TDM

// ---------------------------------------------------------------------------
// Host launcher
// ---------------------------------------------------------------------------
extern "C" void kernel_launch(void* const* d_in, const int* in_sizes, int n_in,
                              void* d_out, int out_size, void* d_ws, size_t ws_size,
                              hipStream_t stream) {
    const float* X     = (const float*)d_in[0];   // [B,T,HID]
    const float* amask = (const float*)d_in[1];   // [B,1,1,T]
    const float* Wq    = (const float*)d_in[2];
    const float* bq    = (const float*)d_in[3];
    const float* Wk    = (const float*)d_in[4];
    const float* bk    = (const float*)d_in[5];
    const float* Wv    = (const float*)d_in[6];
    const float* bv    = (const float*)d_in[7];
    float* out = (float*)d_out;

    unsigned short* Xb  = (unsigned short*)d_ws;                    // 8192*1024
    unsigned short* Wqb = Xb  + (size_t)MROWS * HID;
    unsigned short* Wkb = Wqb + (size_t)HID * HID;
    unsigned short* Wvb = Wkb + (size_t)HID * HID;
    unsigned short* Qb  = Wvb + (size_t)HID * HID;                  // [B,NH,T,HD]
    unsigned short* Kb  = Qb  + (size_t)BB * NH * TT * HD;
    unsigned short* Vtb = Kb  + (size_t)BB * NH * TT * HD;          // [B,NH,HD,T]

    // fp32 -> bf16 converts
    {
        int n4 = (MROWS * HID) / 4;
        cvt_bf16_kernel<<<(n4 + 255) / 256, 256, 0, stream>>>(X, Xb, n4);
        int w4 = (HID * HID) / 4;
        cvt_bf16_kernel<<<(w4 + 255) / 256, 256, 0, stream>>>(Wq, Wqb, w4);
        cvt_bf16_kernel<<<(w4 + 255) / 256, 256, 0, stream>>>(Wk, Wkb, w4);
        cvt_bf16_kernel<<<(w4 + 255) / 256, 256, 0, stream>>>(Wv, Wvb, w4);
    }

    // fused QKV projection
    {
        dim3 grid(MROWS / 64, HID / 64, 3);
        qkv_proj_kernel<<<grid, 128, 0, stream>>>(Xb, Wqb, Wkb, Wvb,
                                                  bq, bk, bv, Qb, Kb, Vtb);
    }

    // flash attention
    {
        dim3 grid(TT / 64, BB * NH);
        attn_kernel<<<grid, 128, 0, stream>>>(Qb, Kb, Vtb, amask, out);
    }
}
